// ConversationScorerModel_58557584114019
// MI455X (gfx1250) — compile-verified, matrix operand order checked by reading, not run on hardware
//
#include <hip/hip_runtime.h>
#include <hip/hip_bf16.h>

typedef unsigned short bf16_t;
typedef unsigned int u32;

typedef __attribute__((ext_vector_type(16))) __bf16 v16bf;
typedef __attribute__((ext_vector_type(8))) float v8f;
typedef __attribute__((ext_vector_type(4))) int v4i_t;

#define DMODEL 768
#define FFDIM 3072
#define SEQLEN 2048
#define NLAYER 12
#define NHEAD 12
#define HEADD 64
#define NBLK 8          // 2048 / 256
#define WIN 256

// -------------------------------------------------- async global->LDS path

#if defined(__has_builtin)
#if __has_builtin(__builtin_amdgcn_global_load_async_to_lds_b128)
#define HAVE_ASYNC 1
#endif
#endif
#ifndef HAVE_ASYNC
#define HAVE_ASYNC 0
#endif

typedef __attribute__((address_space(1))) v4i_t* g_v4i_p;
typedef __attribute__((address_space(3))) v4i_t* l_v4i_p;

__device__ __forceinline__ void async_copy16(void* lds, const void* g) {
#if HAVE_ASYNC
  __builtin_amdgcn_global_load_async_to_lds_b128(
      (g_v4i_p)(unsigned long long)g,
      (l_v4i_p)(unsigned int)(unsigned long long)lds,
      0, 0);
#else
  *(uint4*)lds = *(const uint4*)g;
#endif
}

__device__ __forceinline__ void async_join() {
#if HAVE_ASYNC
#if __has_builtin(__builtin_amdgcn_s_wait_asynccnt)
  __builtin_amdgcn_s_wait_asynccnt(0);
#else
  asm volatile("s_wait_asynccnt 0" ::: "memory");
#endif
#endif
}

// ---------------------------------------------------------------- helpers

__device__ __forceinline__ bf16_t f2bf(float x) {
  u32 u = __float_as_uint(x);
  u32 r = u + 0x7FFFu + ((u >> 16) & 1u);
  return (bf16_t)(r >> 16);
}

// K offset inside a 16-bit WMMA A/B fragment (ISA 7.12.2):
// lanes 0-15: VGPR j<4 -> K=2j,2j+1 ; j>=4 -> K=2j+8 ; lanes 16-31: +8
__device__ __forceinline__ int koff(int j, int lane) {
  return 2 * j + ((j >= 4) ? 8 : 0) + ((lane & 16) ? 8 : 0);
}

// Load a 16x32 (16-bit) WMMA fragment whose K dim is contiguous at p.
__device__ __forceinline__ v16bf load_frag(const bf16_t* p, int lane) {
  union { v16bf v; u32 u[8]; } f;
#pragma unroll
  for (int j = 0; j < 8; ++j)
    f.u[j] = *(const u32*)(p + koff(j, lane));
  return f.v;
}

__device__ __forceinline__ v8f wmma_bf16(v16bf a, v16bf b, v8f c) {
  return __builtin_amdgcn_wmma_f32_16x16x32_bf16(false, a, false, b,
                                                 (short)0, c, false, false);
}

__device__ __forceinline__ float block_sum(float v, float* red) {
  int tid = threadIdx.x;
  __syncthreads();
  red[tid] = v;
  __syncthreads();
  for (int s = blockDim.x >> 1; s > 0; s >>= 1) {
    if (tid < s) red[tid] += red[tid + s];
    __syncthreads();
  }
  return red[0];
}

// ---------------------------------------------------------------- elementwise

__global__ void cast_scale_kernel(const float* __restrict__ in,
                                  bf16_t* __restrict__ out, int n, float scale) {
  int i = blockIdx.x * blockDim.x + threadIdx.x;
  if (i < n) out[i] = f2bf(in[i] * scale);
}

__global__ void gelu_cast_kernel(const float* __restrict__ in,
                                 bf16_t* __restrict__ out, int n) {
  int i = blockIdx.x * blockDim.x + threadIdx.x;
  if (i < n) {
    float x = in[i];
    float g = 0.5f * x * (1.0f + erff(x * 0.70710678118654752f));
    out[i] = f2bf(g);
  }
}

// x = word_emb[id] + pos_emb + tok_emb ; LN ; write fp32 h and bf16 hb
__global__ void embed_ln_kernel(const float* __restrict__ we,
                                const float* __restrict__ pe,
                                const float* __restrict__ te,
                                const float* __restrict__ g,
                                const float* __restrict__ bb,
                                const int* __restrict__ ids,
                                float* __restrict__ h, bf16_t* __restrict__ hb) {
  __shared__ float red[256];
  int s = blockIdx.x, tid = threadIdx.x;
  long id = ids[s];
  float x[3], sum = 0.f, sq = 0.f;
#pragma unroll
  for (int i = 0; i < 3; ++i) {
    int d = tid + i * 256;
    float v = we[id * DMODEL + d] + pe[(long)s * DMODEL + d] + te[d];
    x[i] = v; sum += v; sq += v * v;
  }
  float mean = block_sum(sum, red) * (1.0f / DMODEL);
  float var = block_sum(sq, red) * (1.0f / DMODEL) - mean * mean;
  float inv = rsqrtf(var + 1e-5f);
#pragma unroll
  for (int i = 0; i < 3; ++i) {
    int d = tid + i * 256;
    float o = (x[i] - mean) * inv * g[d] + bb[d];
    h[(long)s * DMODEL + d] = o;
    hb[(long)s * DMODEL + d] = f2bf(o);
  }
}

// h = LN(h + gin) ; writes fp32 h (in place) and bf16 hb
__global__ void add_ln_kernel(float* __restrict__ h,
                              const float* __restrict__ gin,
                              const float* __restrict__ g,
                              const float* __restrict__ bb,
                              bf16_t* __restrict__ hb) {
  __shared__ float red[256];
  int s = blockIdx.x, tid = threadIdx.x;
  float x[3], sum = 0.f, sq = 0.f;
#pragma unroll
  for (int i = 0; i < 3; ++i) {
    int d = tid + i * 256;
    float v = h[(long)s * DMODEL + d] + gin[(long)s * DMODEL + d];
    x[i] = v; sum += v; sq += v * v;
  }
  float mean = block_sum(sum, red) * (1.0f / DMODEL);
  float var = block_sum(sq, red) * (1.0f / DMODEL) - mean * mean;
  float inv = rsqrtf(var + 1e-5f);
#pragma unroll
  for (int i = 0; i < 3; ++i) {
    int d = tid + i * 256;
    float o = (x[i] - mean) * inv * g[d] + bb[d];
    h[(long)s * DMODEL + d] = o;
    hb[(long)s * DMODEL + d] = f2bf(o);
  }
}

// ---------------------------------------------------------------- WMMA GEMM
// C[M,N] fp32 = A[M,K] bf16 x B[K,N] bf16 + bias.
// Block tile 128x128, 8 waves (4 m x 2 n), wave tile 32x64 = 2x4 WMMA tiles.
// Double-buffered LDS; A tile via async global->LDS, B tile register-staged
// and stored transposed (Bt[n][k]) so fragment K is LDS-contiguous.

__global__ __launch_bounds__(256) void gemm_bf16_kernel(
    const bf16_t* __restrict__ A, const bf16_t* __restrict__ B,
    const float* __restrict__ bias, float* __restrict__ C, int N, int K) {
  __shared__ alignas(16) bf16_t As[2][128 * 32];
  __shared__ alignas(16) bf16_t Bt[2][128 * 32];

  const int tid = threadIdx.x;
  const int lane = tid & 31, wave = tid >> 5;
  const int waveM = wave >> 1, waveN = wave & 1;
  const int bm = blockIdx.y * 128, bn = blockIdx.x * 128;

  v8f acc[2][4] = {};

  auto loadA_async = [&](int k0, int buf) {
#pragma unroll
    for (int i = 0; i < 2; ++i) {
      int slot = tid + i * 256;          // 128 rows x 4 segs
      int r = slot >> 2, seg = slot & 3;
      async_copy16(&As[buf][r * 32 + seg * 8],
                   A + (long)(bm + r) * K + k0 + seg * 8);
    }
  };
  auto loadB_regs = [&](int k0, uint4* d) {
#pragma unroll
    for (int i = 0; i < 2; ++i) {
      int slot = tid + i * 256;          // 32 k x 16 segs
      int kk = slot >> 4, seg = slot & 15;
      d[i] = *(const uint4*)(B + (long)(k0 + kk) * N + bn + seg * 8);
    }
  };
  auto storeB = [&](const uint4* d, int buf) {
#pragma unroll
    for (int i = 0; i < 2; ++i) {
      int slot = tid + i * 256;
      int kk = slot >> 4, seg = slot & 15;
      const bf16_t* e = (const bf16_t*)&d[i];
#pragma unroll
      for (int t = 0; t < 8; ++t)
        Bt[buf][(seg * 8 + t) * 32 + kk] = e[t];
    }
  };

  // prologue: tile 0
  uint4 breg[2];
  loadA_async(0, 0);
  loadB_regs(0, breg);
  storeB(breg, 0);
  async_join();
  __syncthreads();

  int cur = 0;
  for (int k0 = 0; k0 < K; k0 += 32) {
    const int nxt = cur ^ 1;
    const bool more = (k0 + 32) < K;
    if (more) {
      loadA_async(k0 + 32, nxt);
      loadB_regs(k0 + 32, breg);
    }

    v16bf a[2], bf_[4];
#pragma unroll
    for (int mt = 0; mt < 2; ++mt)
      a[mt] = load_frag(&As[cur][(waveM * 32 + mt * 16 + (lane & 15)) * 32], lane);
#pragma unroll
    for (int nt = 0; nt < 4; ++nt)
      bf_[nt] = load_frag(&Bt[cur][(waveN * 64 + nt * 16 + (lane & 15)) * 32], lane);

#pragma unroll
    for (int mt = 0; mt < 2; ++mt)
#pragma unroll
      for (int nt = 0; nt < 4; ++nt)
        acc[mt][nt] = wmma_bf16(a[mt], bf_[nt], acc[mt][nt]);

    if (more) storeB(breg, nxt);
    async_join();
    __syncthreads();
    cur = nxt;
  }

  const int halfm = (lane >> 4) << 3;
  const int cn = lane & 15;
#pragma unroll
  for (int mt = 0; mt < 2; ++mt)
#pragma unroll
    for (int nt = 0; nt < 4; ++nt) {
      int col = bn + waveN * 64 + nt * 16 + cn;
      float bv = bias ? bias[col] : 0.f;
#pragma unroll
      for (int r = 0; r < 8; ++r) {
        int row = bm + waveM * 32 + mt * 16 + halfm + r;
        C[(long)row * N + col] = acc[mt][nt][r] + bv;
      }
    }
}

// ---------------------------------------------------------------- attention
// One workgroup per (head, q-block). 8 waves, 32 q-rows each. Streams the
// 768-wide banded key range in 64-key chunks with online softmax; QK^T and
// PV both via WMMA bf16. Q is pre-scaled by hd^-0.5.

__global__ __launch_bounds__(256) void attn_kernel(
    const bf16_t* __restrict__ Q, const bf16_t* __restrict__ Kb,
    const bf16_t* __restrict__ Vb, const int* __restrict__ amask,
    float* __restrict__ O) {
  __shared__ alignas(16) bf16_t Ks[64 * 64];      // [key][d]
  __shared__ alignas(16) bf16_t Vt[64 * 64];      // [d][key]
  __shared__ alignas(16) bf16_t Ps[8][16 * 64];   // per-wave P tile
  __shared__ int kflag[64];

  const int h = blockIdx.x;   // head
  const int b = blockIdx.y;   // query block
  const int tid = threadIdx.x, lane = tid & 31, w = tid >> 5;
  const int qrow0 = b * WIN + w * 32;

  // Q fragments: rows are K-contiguous in global (d, d+1 adjacent)
  v16bf aq[2][2];
#pragma unroll
  for (int mt = 0; mt < 2; ++mt)
#pragma unroll
    for (int kc = 0; kc < 2; ++kc)
      aq[mt][kc] = load_frag(
          Q + (long)(qrow0 + mt * 16 + (lane & 15)) * DMODEL + h * HEADD + kc * 32,
          lane);

  v8f o[2][4] = {};
  float mstate[2][8], lstate[2][8];
#pragma unroll
  for (int mt = 0; mt < 2; ++mt)
#pragma unroll
    for (int r = 0; r < 8; ++r) { mstate[mt][r] = -1e30f; lstate[mt][r] = 0.f; }

  const int halfm = (lane >> 4) << 3;
  const int cn = lane & 15;

  for (int c = 0; c < 12; ++c) {   // 12 chunks of 64 keys = 768 = 3W
    __syncthreads();
    if (tid < 64) {
      int skey = b * WIN + c * 64 + tid - WIN;
      kflag[tid] = (skey >= 0 && skey < SEQLEN) ? (amask[skey] != 0) : 0;
    }
#pragma unroll
    for (int i = 0; i < 2; ++i) {
      int slot = tid + i * 256;          // 64 keys x 8 segs
      int j = slot >> 3, seg = slot & 7;
      int skey = b * WIN + c * 64 + j - WIN;
      bool val = (skey >= 0 && skey < SEQLEN);
      if (val)
        async_copy16(Ks + j * 64 + seg * 8,
                     Kb + (long)skey * DMODEL + h * HEADD + seg * 8);
      else
        *(uint4*)(Ks + j * 64 + seg * 8) = make_uint4(0u, 0u, 0u, 0u);
      uint4 vd = make_uint4(0u, 0u, 0u, 0u);
      if (val)
        vd = *(const uint4*)(Vb + (long)skey * DMODEL + h * HEADD + seg * 8);
      const bf16_t* ve = (const bf16_t*)&vd;
#pragma unroll
      for (int t = 0; t < 8; ++t)
        Vt[(seg * 8 + t) * 64 + j] = ve[t];
    }
    async_join();
    __syncthreads();

#pragma unroll
    for (int mt = 0; mt < 2; ++mt) {
      // ---- scores: S = Q Kt  (contraction over d = 64 -> 2 wmma steps)
      v8f s[4] = {};
#pragma unroll
      for (int kc = 0; kc < 2; ++kc)
#pragma unroll
        for (int nt = 0; nt < 4; ++nt) {
          v16bf bk = load_frag(Ks + (nt * 16 + cn) * 64 + kc * 32, lane);
          s[nt] = wmma_bf16(aq[mt][kc], bk, s[nt]);
        }

      int fl[4];
#pragma unroll
      for (int nt = 0; nt < 4; ++nt) fl[nt] = kflag[nt * 16 + cn];

#pragma unroll
      for (int r = 0; r < 8; ++r) {
        int qq = w * 32 + mt * 16 + halfm + r;       // 0..255 in block
        float cm = -1e30f;
#pragma unroll
        for (int nt = 0; nt < 4; ++nt) {
          int j = c * 64 + nt * 16 + cn;             // 0..767 in k3
          bool band = (j >= qq) && (j <= qq + 2 * WIN);
          float x = (band && fl[nt]) ? s[nt][r] : -1e9f;
          s[nt][r] = x;
          cm = fmaxf(cm, x);
        }
#pragma unroll
        for (int off = 1; off < 16; off <<= 1)
          cm = fmaxf(cm, __shfl_xor(cm, off, 32));
        float mold = mstate[mt][r];
        float mnew = fmaxf(mold, cm);
        float scale = __expf(mold - mnew);
        float rs = 0.f;
#pragma unroll
        for (int nt = 0; nt < 4; ++nt) {
          float p = __expf(s[nt][r] - mnew);
          rs += p;
          Ps[w][(halfm + r) * 64 + nt * 16 + cn] = f2bf(p);
        }
#pragma unroll
        for (int off = 1; off < 16; off <<= 1)
          rs += __shfl_xor(rs, off, 32);
        lstate[mt][r] = lstate[mt][r] * scale + rs;
        mstate[mt][r] = mnew;
#pragma unroll
        for (int dt = 0; dt < 4; ++dt) o[mt][dt][r] *= scale;
      }

      // ---- O += P V  (contraction over 64 keys -> 2 wmma steps)
#pragma unroll
      for (int kc2 = 0; kc2 < 2; ++kc2) {
        v16bf ap = load_frag(&Ps[w][cn * 64 + kc2 * 32], lane);
#pragma unroll
        for (int dt = 0; dt < 4; ++dt) {
          v16bf bv = load_frag(Vt + (dt * 16 + cn) * 64 + kc2 * 32, lane);
          o[mt][dt] = wmma_bf16(ap, bv, o[mt][dt]);
        }
      }
    }
  }

#pragma unroll
  for (int mt = 0; mt < 2; ++mt)
#pragma unroll
    for (int r = 0; r < 8; ++r) {
      float inv = 1.0f / lstate[mt][r];
      int row = qrow0 + mt * 16 + halfm + r;
#pragma unroll
      for (int dt = 0; dt < 4; ++dt)
        O[(long)row * DMODEL + h * HEADD + dt * 16 + cn] = o[mt][dt][r] * inv;
    }
}

// ---------------------------------------------------------------- tail

__global__ void pool_logit_kernel(const float* __restrict__ h,
                                  const float* __restrict__ W1,
                                  const float* __restrict__ b1,
                                  const float* __restrict__ W2,
                                  const float* __restrict__ b2,
                                  const int* __restrict__ amask,
                                  float* __restrict__ logits) {
  __shared__ float red[64];
  int s = blockIdx.x, j = threadIdx.x;
  float acc = b1[j];
  for (int i = 0; i < DMODEL; ++i)
    acc += h[(long)s * DMODEL + i] * W1[i * 64 + j];
  float p = tanhf(acc) * W2[j];
  red[j] = p;
  __syncthreads();
  for (int st = 32; st > 0; st >>= 1) {
    if (j < st) red[j] += red[j + st];
    __syncthreads();
  }
  if (j == 0) {
    float l = red[0] + b2[0];
    logits[s] = amask[s] ? l : -1e9f;
  }
}

__global__ void softmax_aw_kernel(const float* __restrict__ logits,
                                  const int* __restrict__ amask,
                                  float* __restrict__ aw,
                                  float* __restrict__ scal) {
  __shared__ float red[256];
  int tid = threadIdx.x;
  float mx = -1e30f;
  for (int s = tid; s < SEQLEN; s += 256) mx = fmaxf(mx, logits[s]);
  __syncthreads();
  red[tid] = mx; __syncthreads();
  for (int st = 128; st > 0; st >>= 1) {
    if (tid < st) red[tid] = fmaxf(red[tid], red[tid + st]);
    __syncthreads();
  }
  mx = red[0];
  float sum = 0.f;
  for (int s = tid; s < SEQLEN; s += 256) {
    float e = __expf(logits[s] - mx);
    aw[s] = e; sum += e;
  }
  sum = block_sum(sum, red);
  for (int s = tid; s < SEQLEN; s += 256) aw[s] /= sum;
  float ms = 0.f;
  for (int s = tid; s < SEQLEN; s += 256) ms += (float)amask[s];
  ms = block_sum(ms, red);
  if (tid == 0) scal[0] = ms;
}

__global__ void pooled_kernel(const float* __restrict__ h,
                              const float* __restrict__ aw,
                              const int* __restrict__ amask,
                              const float* __restrict__ scal,
                              float* __restrict__ pooled) {
  int d = blockIdx.x * blockDim.x + threadIdx.x;
  if (d >= DMODEL) return;
  float aacc = 0.f, macc = 0.f;
  for (int s = 0; s < SEQLEN; ++s) {
    float v = h[(long)s * DMODEL + d];
    aacc += v * aw[s];
    macc += v * (float)amask[s];
  }
  pooled[d] = aacc;
  pooled[DMODEL + d] = macc / fmaxf(scal[0], 1e-9f);
}

__global__ void matvec_kernel(const float* __restrict__ x,
                              const float* __restrict__ W,
                              const float* __restrict__ bias,
                              float* __restrict__ y, int In, int Out) {
  int j = blockIdx.x * blockDim.x + threadIdx.x;
  if (j >= Out) return;
  float acc = bias[j];
  for (int i = 0; i < In; ++i) acc += x[i] * W[(long)i * Out + j];
  y[j] = acc;
}

// act: 0 none, 1 gelu, 2 relu (applied after LN)
__global__ void ln_act_kernel(const float* __restrict__ in,
                              const float* __restrict__ g,
                              const float* __restrict__ bb,
                              float* __restrict__ out, int n, int act) {
  __shared__ float red[256];
  int tid = threadIdx.x;
  float sum = 0.f, sq = 0.f;
  for (int i = tid; i < n; i += 256) { float v = in[i]; sum += v; sq += v * v; }
  float mean = block_sum(sum, red) / n;
  float var = block_sum(sq, red) / n - mean * mean;
  float inv = rsqrtf(var + 1e-5f);
  for (int i = tid; i < n; i += 256) {
    float o = (in[i] - mean) * inv * g[i] + bb[i];
    if (act == 1) o = 0.5f * o * (1.0f + erff(o * 0.70710678118654752f));
    else if (act == 2) o = fmaxf(o, 0.f);
    out[i] = o;
  }
}

__global__ void head_kernel(const float* __restrict__ x,
                            const float* __restrict__ W,
                            const float* __restrict__ bias,
                            float* __restrict__ out) {
  __shared__ float red[256];
  int tid = threadIdx.x;
  for (int o = 0; o < 5; ++o) {
    float p = 0.f;
    for (int d = tid; d < DMODEL; d += 256) p += x[d] * W[d * 5 + o];
    float s = block_sum(p, red);
    if (tid == 0) out[o] = s + bias[o];
  }
}

// ---------------------------------------------------------------- launch

extern "C" void kernel_launch(void* const* d_in, const int* in_sizes, int n_in,
                              void* d_out, int out_size, void* d_ws, size_t ws_size,
                              hipStream_t stream) {
  // params in setup_inputs() order
  const float* word_emb = (const float*)d_in[0];
  const float* pos_emb  = (const float*)d_in[1];
  const float* tok_emb  = (const float*)d_in[2];
  const float* emb_g    = (const float*)d_in[3];
  const float* emb_b    = (const float*)d_in[4];
  const float* Wq = (const float*)d_in[5];  const float* bq = (const float*)d_in[6];
  const float* Wk = (const float*)d_in[7];  const float* bk = (const float*)d_in[8];
  const float* Wv = (const float*)d_in[9];  const float* bv = (const float*)d_in[10];
  const float* Wo = (const float*)d_in[11]; const float* bo = (const float*)d_in[12];
  const float* ln1g = (const float*)d_in[13]; const float* ln1b = (const float*)d_in[14];
  const float* Wi = (const float*)d_in[15]; const float* bi = (const float*)d_in[16];
  const float* Wf = (const float*)d_in[17]; const float* bf = (const float*)d_in[18];
  const float* ln2g = (const float*)d_in[19]; const float* ln2b = (const float*)d_in[20];
  const float* pool_W1 = (const float*)d_in[21]; const float* pool_b1 = (const float*)d_in[22];
  const float* pool_W2 = (const float*)d_in[23]; const float* pool_b2 = (const float*)d_in[24];
  const float* fc_W1 = (const float*)d_in[25]; const float* fc_b1 = (const float*)d_in[26];
  const float* fc_ln1g = (const float*)d_in[27]; const float* fc_ln1b = (const float*)d_in[28];
  const float* fc_W2 = (const float*)d_in[29]; const float* fc_b2 = (const float*)d_in[30];
  const float* fc_ln2g = (const float*)d_in[31]; const float* fc_ln2b = (const float*)d_in[32];
  const float* head_W = (const float*)d_in[33]; const float* head_b = (const float*)d_in[34];
  const int* ids   = (const int*)d_in[35];
  const int* amask = (const int*)d_in[36];

  // workspace carve-up
  size_t off = 0;
  auto alloc = [&](size_t bytes) -> void* {
    off = (off + 255) & ~(size_t)255;
    void* p = (char*)d_ws + off;
    off += bytes;
    return p;
  };
  float* h   = (float*)alloc((size_t)SEQLEN * DMODEL * 4);
  float* y   = (float*)alloc((size_t)SEQLEN * DMODEL * 4);
  float* ff  = (float*)alloc((size_t)SEQLEN * FFDIM * 4);
  bf16_t* hb  = (bf16_t*)alloc((size_t)SEQLEN * DMODEL * 2);
  bf16_t* qb  = (bf16_t*)alloc((size_t)SEQLEN * DMODEL * 2);
  bf16_t* kb  = (bf16_t*)alloc((size_t)SEQLEN * DMODEL * 2);
  bf16_t* vb  = (bf16_t*)alloc((size_t)SEQLEN * DMODEL * 2);
  bf16_t* ffb = (bf16_t*)alloc((size_t)SEQLEN * FFDIM * 2);
  bf16_t* wslab = (bf16_t*)alloc((size_t)DMODEL * FFDIM * 2);
  float* logits = (float*)alloc(SEQLEN * 4);
  float* aw     = (float*)alloc(SEQLEN * 4);
  float* scal   = (float*)alloc(16);
  float* pooled = (float*)alloc(2 * DMODEL * 4);
  float* t1     = (float*)alloc(2 * DMODEL * 4);
  float* t2     = (float*)alloc(2 * DMODEL * 4);
  float* t3     = (float*)alloc(DMODEL * 4);

  const int DD = DMODEL * DMODEL;
  const int DF = DMODEL * FFDIM;
  const int SD = SEQLEN * DMODEL;
  const int SF = SEQLEN * FFDIM;

  auto conv = [&](const float* w, int n) {
    cast_scale_kernel<<<(n + 255) / 256, 256, 0, stream>>>(w, wslab, n, 1.0f);
  };
  auto gemm = [&](const bf16_t* A, const float* bias, float* C, int N, int K) {
    gemm_bf16_kernel<<<dim3(N / 128, SEQLEN / 128), 256, 0, stream>>>(
        A, wslab, bias, C, N, K);
  };

  embed_ln_kernel<<<SEQLEN, 256, 0, stream>>>(word_emb, pos_emb, tok_emb,
                                              emb_g, emb_b, ids, h, hb);

  for (int l = 0; l < NLAYER; ++l) {
    // Q/K/V projections (bf16 WMMA), Q pre-scaled by hd^-0.5
    conv(Wq + (size_t)l * DD, DD);
    gemm(hb, bq + (size_t)l * DMODEL, y, DMODEL, DMODEL);
    cast_scale_kernel<<<(SD + 255) / 256, 256, 0, stream>>>(y, qb, SD, 0.125f);
    conv(Wk + (size_t)l * DD, DD);
    gemm(hb, bk + (size_t)l * DMODEL, y, DMODEL, DMODEL);
    cast_scale_kernel<<<(SD + 255) / 256, 256, 0, stream>>>(y, kb, SD, 1.0f);
    conv(Wv + (size_t)l * DD, DD);
    gemm(hb, bv + (size_t)l * DMODEL, y, DMODEL, DMODEL);
    cast_scale_kernel<<<(SD + 255) / 256, 256, 0, stream>>>(y, vb, SD, 1.0f);

    // sliding-window attention (flash-style, WMMA)
    attn_kernel<<<dim3(NHEAD, NBLK), 256, 0, stream>>>(qb, kb, vb, amask, y);

    // output projection + residual + LN1
    cast_scale_kernel<<<(SD + 255) / 256, 256, 0, stream>>>(y, qb, SD, 1.0f);
    conv(Wo + (size_t)l * DD, DD);
    gemm(qb, bo + (size_t)l * DMODEL, ff, DMODEL, DMODEL);  // reuse ff as tmp
    add_ln_kernel<<<SEQLEN, 256, 0, stream>>>(h, ff, ln1g + (size_t)l * DMODEL,
                                              ln1b + (size_t)l * DMODEL, hb);

    // FFN
    conv(Wi + (size_t)l * DF, DF);
    gemm(hb, bi + (size_t)l * FFDIM, ff, FFDIM, DMODEL);
    gelu_cast_kernel<<<(SF + 255) / 256, 256, 0, stream>>>(ff, ffb, SF);
    conv(Wf + (size_t)l * DF, DF);
    gemm(ffb, bf + (size_t)l * DMODEL, y, DMODEL, FFDIM);
    add_ln_kernel<<<SEQLEN, 256, 0, stream>>>(h, y, ln2g + (size_t)l * DMODEL,
                                              ln2b + (size_t)l * DMODEL, hb);
  }

  // pooling + MLP head
  pool_logit_kernel<<<SEQLEN, 64, 0, stream>>>(h, pool_W1, pool_b1, pool_W2,
                                               pool_b2, amask, logits);
  softmax_aw_kernel<<<1, 256, 0, stream>>>(logits, amask, aw, scal);
  pooled_kernel<<<(DMODEL + 127) / 128, 128, 0, stream>>>(h, aw, amask, scal, pooled);
  matvec_kernel<<<(2 * DMODEL + 127) / 128, 128, 0, stream>>>(pooled, fc_W1, fc_b1,
                                                              t1, 2 * DMODEL, 2 * DMODEL);
  ln_act_kernel<<<1, 256, 0, stream>>>(t1, fc_ln1g, fc_ln1b, t2, 2 * DMODEL, 1);
  matvec_kernel<<<(DMODEL + 127) / 128, 128, 0, stream>>>(t2, fc_W2, fc_b2,
                                                          t1, 2 * DMODEL, DMODEL);
  ln_act_kernel<<<1, 256, 0, stream>>>(t1, fc_ln2g, fc_ln2b, t3, DMODEL, 2);
  head_kernel<<<1, 256, 0, stream>>>(t3, head_W, head_b, (float*)d_out);
}
